// RegimeConditionedGraphAttention_80427557585566
// MI455X (gfx1250) — matrix operand order repaired and checked by reading, not compiled, not run on hardware
//
#include <hip/hip_runtime.h>
#include <hip/hip_bf16.h>

typedef __attribute__((ext_vector_type(16))) __bf16   v16bf;
typedef __attribute__((ext_vector_type(8)))  __bf16   v8bf;
typedef __attribute__((ext_vector_type(8)))  float    v8f;
typedef __attribute__((ext_vector_type(4)))  unsigned u32x4;
typedef __attribute__((ext_vector_type(8)))  int      i32x8;
typedef __attribute__((ext_vector_type(4)))  int      i32x4;

#define N_STOCK 4096
#define DIM     512
#define NHEAD   8
#define HDIM    64
#define KKEEP   2048
#define SC_LD   (NHEAD * N_STOCK)   // 32768
#define NM      (N_STOCK + 2)       // 4098

#if __has_builtin(__builtin_amdgcn_tensor_load_to_lds)
#define HAVE_TDM 1
#else
#define HAVE_TDM 0
#endif

// ---------------------------------------------------------------------------
// TDM: DMA a 64x64 bf16 tile (row stride = strideElems) from global into LDS.
// D# per CDNA5 ISA 8.3/8.4: group0 = {count=1, lds_addr, ga_lo, ga_hi|type=2};
// group1 = {data_size=2B, tensor_dim0/1 = 2^30 (no OOB), tile=64x64,
// tensor_dim0_stride = strideElems}.  Wave-level (EXEC ignored), TENSORcnt.
// This toolchain's builtin: 6 args (g0, g1, g2, g3, g_extra8, cpol).
// ---------------------------------------------------------------------------
__device__ __forceinline__ void tdm_load_tile64(const void* gaddr, unsigned lds_off,
                                                int strideElems)
{
#if HAVE_TDM
    unsigned long long ga = (unsigned long long)gaddr;
    u32x4 g0 = { 1u,                                     // count=1, user mode
                 lds_off,                                // lds_addr (bytes)
                 (unsigned)(ga & 0xFFFFFFFFu),           // global_addr[31:0]
                 ((unsigned)((ga >> 32) & 0x01FFFFFFu))  // global_addr[56:32]
                 | 0x80000000u };                        // type=2 (bits 127:126)
    i32x8 g1 = { 0x00010000,            // data_size=1 (2 bytes), mask=0
                 0,                     // atomic barrier addr / dim0 lo16 = 0
                 0x4000,                // tensor_dim0 = 1<<30 (hi16), dim1 lo16=0
                 (64 << 16) | 0x4000,   // tile_dim0=64 | tensor_dim1 hi16 (1<<30)
                 64,                    // tile_dim1=64, tile_dim2=0
                 strideElems,           // tensor_dim0_stride[31:0]
                 0, 0 };
    i32x4 gz4 = { 0, 0, 0, 0 };
    i32x8 gz8 = { 0, 0, 0, 0, 0, 0, 0, 0 };
    __builtin_amdgcn_tensor_load_to_lds(g0, g1, gz4, gz4, gz8, 0);
#else
    (void)gaddr; (void)lds_off; (void)strideElems;
#endif
}

__device__ __forceinline__ void tdm_wait()
{
#if HAVE_TDM
    __builtin_amdgcn_s_wait_tensorcnt(0);
#endif
}

// 16-B aligned LDS vector load of two 8-elem chunks -> one 16-elem fragment
__device__ __forceinline__ v16bf ldfrag(const __bf16* p0, const __bf16* p1)
{
    v8bf lo = *(const v8bf*)p0;
    v8bf hi = *(const v8bf*)p1;
    return __builtin_shufflevector(lo, hi, 0, 1, 2, 3, 4, 5, 6, 7,
                                           8, 9, 10, 11, 12, 13, 14, 15);
}

// ---------------------------------------------------------------------------
// Templated 4-wave WMMA bf16 GEMM:  C[M x Nn] = A[M x K] * op(B)  (fp32 acc)
//   Block = 128 thr = 4 waves (2x2); each wave owns a 32x32 C macro-tile
//   (2 A-frags x 2 B-frags -> 4 WMMAs / 32-k half-step, 8 per stage).
//   Double-buffered 64x64 LDS tiles; A (and B when NN/bf16) staged by TDM,
//   overlapped with WMMA on the other buffer.
//   MODE: 0 f32 | 1 bf16*regime | 2 bf16 | 3 +bias f32+bf16 | 4 sigmoid f32
//         5 bf16 + transposed bf16 copy (for K -> KbT)
// ---------------------------------------------------------------------------
template<int MODE>
__device__ __forceinline__ void store_elem(float v, long cidx, int ncol, int grow,
                                           float* __restrict__ Cf,
                                           __bf16* __restrict__ Cb,
                                           __bf16* __restrict__ CbT, int ldct,
                                           const float* __restrict__ epi)
{
    if constexpr (MODE == 0)      Cf[cidx] = v;
    else if constexpr (MODE == 1) Cb[cidx] = (__bf16)(v * epi[ncol]);
    else if constexpr (MODE == 2) Cb[cidx] = (__bf16)v;
    else if constexpr (MODE == 3) { float g = v + epi[ncol];
                                    Cf[cidx] = g; Cb[cidx] = (__bf16)g; }
    else if constexpr (MODE == 4) Cf[cidx] = 1.f / (1.f + __expf(-(v + epi[ncol])));
    else { __bf16 b = (__bf16)v; Cb[cidx] = b; CbT[(long)ncol * ldct + grow] = b; }
}

template<int MODE, bool TRANSB, bool BF32, bool SPLIT>
__global__ void __launch_bounds__(128)
wmma_gemm(const __bf16* __restrict__ A0, int lda0,
          const __bf16* __restrict__ A1, int lda1, int kSplit,
          const void* __restrict__ Bv, int ldb,
          float* __restrict__ Cf, __bf16* __restrict__ Cb, int ldc,
          __bf16* __restrict__ CbT, int ldct,
          int K, const float* __restrict__ epi, float alpha,
          long hA, long hB, long hC)
{
    __shared__ __bf16 As[2][64][64];     // [buf][m][k]
    __shared__ __bf16 Bs[2][64][64];     // [buf][k][n]

    const int tid  = threadIdx.x;
    const int wave = tid >> 5, lane = tid & 31;
    const int wy = wave >> 1, wx = wave & 1;
    const int mBase = blockIdx.y * 64;
    const int nBase = blockIdx.x * 64;
    const long z = blockIdx.z;

    A0 += z * hA;
    const float*  Bf = (const float*)Bv + z * hB;
    const __bf16* Bb = (const __bf16*)Bv + z * hB;
    if (Cf) Cf += z * hC;
    if (Cb) Cb += z * hC;

    constexpr bool kTdm  = (HAVE_TDM != 0);
    constexpr bool kTdmB = kTdm && !TRANSB && !BF32;   // NN bf16 B tile

    // manual staging coords: 32 contiguous elements per thread
    const int arow  = tid >> 1;
    const int acol0 = (tid & 1) * 32;
    const int brow  = tid >> 1;
    const int bcol0 = (tid & 1) * 32;

    v8f acc00 = {}, acc01 = {}, acc10 = {}, acc11 = {};

    auto stage = [&](int kk, int buf) {
        // ---- A tile ----
        if constexpr (kTdm) {
            if (wave == 0) {
                const __bf16* src; int stride;
                if (SPLIT && kk >= kSplit) { src = A1 + (long)mBase * lda1 + (kk - kSplit); stride = lda1; }
                else                       { src = A0 + (long)mBase * lda0 + kk;            stride = lda0; }
                tdm_load_tile64(src, (unsigned)(size_t)&As[buf][0][0], stride);
            }
        } else {
            const __bf16* src;
            int kg = kk + acol0;
            if (SPLIT && kg >= kSplit) src = A1 + (long)(mBase + arow) * lda1 + (kg - kSplit);
            else                       src = A0 + (long)(mBase + arow) * lda0 + kg;
            #pragma unroll
            for (int j = 0; j < 32; ++j) As[buf][arow][acol0 + j] = src[j];
        }
        // ---- B tile ----
        if constexpr (kTdmB) {
            if (wave == 1)
                tdm_load_tile64(Bb + (long)kk * ldb + nBase,
                                (unsigned)(size_t)&Bs[buf][0][0], ldb);
        } else if constexpr (TRANSB) {
            long base = (long)(nBase + brow) * ldb + kk + bcol0;
            #pragma unroll
            for (int j = 0; j < 32; ++j)
                Bs[buf][bcol0 + j][brow] = BF32 ? (__bf16)Bf[base + j] : Bb[base + j];
        } else {
            long base = (long)(kk + brow) * ldb + nBase + bcol0;
            #pragma unroll
            for (int j = 0; j < 32; ++j)
                Bs[buf][brow][bcol0 + j] = BF32 ? (__bf16)Bf[base + j] : Bb[base + j];
        }
    };

    auto compute = [&](int buf) {
        const int mrow = lane & 15;
        const int kb   = (lane >> 4) << 3;
        const int r0 = wy * 32 + mrow, r1 = r0 + 16;
        #pragma unroll
        for (int ks = 0; ks < 64; ks += 32) {
            v16bf a0v = ldfrag(&As[buf][r0][ks + kb],      &As[buf][r0][ks + 16 + kb]);
            v16bf a1v = ldfrag(&As[buf][r1][ks + kb],      &As[buf][r1][ks + 16 + kb]);
            v16bf b0v, b1v;
            #pragma unroll
            for (int s = 0; s < 16; ++s) {
                const int kr = ks + kb + (s & 7) + ((s >> 3) << 4);
                b0v[s] = Bs[buf][kr][wx * 32 + mrow];
                b1v[s] = Bs[buf][kr][wx * 32 + 16 + mrow];
            }
            acc00 = __builtin_amdgcn_wmma_f32_16x16x32_bf16(false, a0v, false, b0v, (short)0, acc00, false, false);
            acc01 = __builtin_amdgcn_wmma_f32_16x16x32_bf16(false, a0v, false, b1v, (short)0, acc01, false, false);
            acc10 = __builtin_amdgcn_wmma_f32_16x16x32_bf16(false, a1v, false, b0v, (short)0, acc10, false, false);
            acc11 = __builtin_amdgcn_wmma_f32_16x16x32_bf16(false, a1v, false, b1v, (short)0, acc11, false, false);
        }
    };

    // prologue: fill buffer 0
    stage(0, 0);
    tdm_wait();
    __syncthreads();

    for (int kk = 0; kk < K; kk += 64) {
        const int cur = (kk >> 6) & 1;
        if (kk + 64 < K) stage(kk + 64, cur ^ 1);   // DMA next while computing
        compute(cur);
        tdm_wait();
        __syncthreads();
    }

    // ---- epilogue ----
    const int nlo  = lane & 15;
    const int mOff = (lane >> 4) * 8;
    #pragma unroll
    for (int t = 0; t < 2; ++t) {
        #pragma unroll
        for (int u = 0; u < 2; ++u) {
            const v8f acc = (t == 0) ? (u == 0 ? acc00 : acc01)
                                     : (u == 0 ? acc10 : acc11);
            const int ncol = nBase + wx * 32 + u * 16 + nlo;
            #pragma unroll
            for (int r = 0; r < 8; ++r) {
                int grow = mBase + wy * 32 + t * 16 + mOff + r;
                store_elem<MODE>(acc[r] * alpha, (long)grow * ldc + ncol, ncol, grow,
                                 Cf, Cb, CbT, ldct, epi);
            }
        }
    }
}

// ---------------------------------------------------------------------------
__global__ void regime_kernel(const float* __restrict__ z1, const float* __restrict__ z2,
                              const float* __restrict__ Wg, const float* __restrict__ bg,
                              float* __restrict__ regime)
{
    int o = blockIdx.x * blockDim.x + threadIdx.x;
    if (o >= DIM) return;
    const float* w = Wg + (long)o * (2 * DIM);
    float s = bg[o];
    for (int i = 0; i < DIM; ++i) s += z1[i] * w[i];
    for (int i = 0; i < DIM; ++i) s += z2[i] * w[DIM + i];
    regime[o] = 1.f / (1.f + __expf(-s));
}

__global__ void cast_bf16_kernel(const float* __restrict__ in, __bf16* __restrict__ out, long n)
{
    long i = (long)blockIdx.x * blockDim.x + threadIdx.x;
    if (i < n) out[i] = (__bf16)in[i];
}

__global__ void anchor_kv_kernel(const float* __restrict__ z1, const float* __restrict__ z2,
                                 const float* __restrict__ Wk, const float* __restrict__ Wv,
                                 float* __restrict__ kA, float* __restrict__ vA)
{
    int t = blockIdx.x * blockDim.x + threadIdx.x;
    if (t >= 2 * DIM) return;
    int a = t >> 9, o = t & (DIM - 1);
    const float* z = a ? z2 : z1;
    float sk = 0.f, sv = 0.f;
    const float* wk = Wk + (long)o * DIM;
    const float* wv = Wv + (long)o * DIM;
    for (int i = 0; i < DIM; ++i) { sk += z[i] * wk[i]; sv += z[i] * wv[i]; }
    kA[a * DIM + o] = sk;
    vA[a * DIM + o] = sv;
}

__global__ void anchor_scores_kernel(const __bf16* __restrict__ Qb,
                                     const float* __restrict__ kA,
                                     float* __restrict__ ascore)
{
    int idx = blockIdx.x * blockDim.x + threadIdx.x;
    if (idx >= N_STOCK * NHEAD * 2) return;
    int n = idx >> 4, rem = idx & 15, h = rem >> 1, a = rem & 1;
    float s = 0.f;
    const __bf16* q = Qb + (long)n * DIM + h * HDIM;
    const float*  k = kA + a * DIM + h * HDIM;
    for (int d = 0; d < HDIM; ++d) s += (float)q[d] * k[d];
    ascore[idx] = s * 0.125f;
}

// ---------------------------------------------------------------------------
__global__ void __launch_bounds__(256)
topk_softmax_kernel(const float* __restrict__ scores, const float* __restrict__ ascore,
                    float* __restrict__ attnA, __bf16* __restrict__ attnS)
{
    const int row = blockIdx.x;
    const int n = row >> 3, h = row & 7;
    const int tid = threadIdx.x;
    const float* srow = scores + (long)n * SC_LD + h * N_STOCK;

    __shared__ unsigned hist[256];
    __shared__ float    red[256];
    __shared__ unsigned s_sel, s_kc;

    float    vals[16];
    unsigned keys[16];
    #pragma unroll
    for (int i = 0; i < 16; ++i) {
        float x = srow[i * 256 + tid];
        vals[i] = x;
        unsigned u = __float_as_uint(x);
        keys[i] = (u & 0x80000000u) ? ~u : (u | 0x80000000u);
    }

    unsigned prefix = 0, kneed = KKEEP;
    for (int pass = 3; pass >= 0; --pass) {
        const int shift = pass * 8;
        hist[tid] = 0;
        __syncthreads();
        const unsigned hiMask = (pass == 3) ? 0u : (0xFFFFFFFFu << (shift + 8));
        #pragma unroll
        for (int i = 0; i < 16; ++i)
            if ((keys[i] & hiMask) == (prefix & hiMask))
                atomicAdd(&hist[(keys[i] >> shift) & 255u], 1u);
        __syncthreads();
        if (tid == 0) {
            unsigned c = 0; int b = 255;
            for (; b >= 0; --b) { c += hist[b]; if (c >= kneed) break; }
            if (b < 0) b = 0;
            s_sel = (unsigned)b;
            s_kc  = kneed - (c - hist[b]);
        }
        __syncthreads();
        prefix |= (s_sel << shift);
        kneed   = s_kc;
    }
    const unsigned tu = prefix;
    const float T = __uint_as_float((tu & 0x80000000u) ? (tu ^ 0x80000000u) : ~tu);

    const float a0 = ascore[n * 16 + h * 2 + 0];
    const float a1 = ascore[n * 16 + h * 2 + 1];

    float lmax = -3.4e38f;
    #pragma unroll
    for (int i = 0; i < 16; ++i) lmax = fmaxf(lmax, vals[i]);
    red[tid] = lmax;
    __syncthreads();
    for (int off = 128; off > 0; off >>= 1) {
        if (tid < off) red[tid] = fmaxf(red[tid], red[tid + off]);
        __syncthreads();
    }
    const float rmax = fmaxf(fmaxf(red[0], a0), a1);
    __syncthreads();

    float lsum = 0.f;
    #pragma unroll
    for (int i = 0; i < 16; ++i) {
        float ex = (vals[i] >= T) ? __expf(vals[i] - rmax) : 0.f;
        vals[i] = ex;
        lsum += ex;
    }
    red[tid] = lsum;
    __syncthreads();
    for (int off = 128; off > 0; off >>= 1) {
        if (tid < off) red[tid] += red[tid + off];
        __syncthreads();
    }
    const float e0 = __expf(a0 - rmax), e1 = __expf(a1 - rmax);
    const float inv = 1.f / (red[0] + e0 + e1);
    if (tid == 0) {
        attnA[n * 16 + h * 2 + 0] = e0 * inv;
        attnA[n * 16 + h * 2 + 1] = e1 * inv;
    }
    __bf16* arow = attnS + (long)n * SC_LD + h * N_STOCK;
    #pragma unroll
    for (int i = 0; i < 16; ++i)
        arow[i * 256 + tid] = (__bf16)(vals[i] * inv);
}

__global__ void rawscores_kernel(const float* __restrict__ scores,
                                 const float* __restrict__ ascore,
                                 float* __restrict__ dst)
{
    long idx = (long)blockIdx.x * blockDim.x + threadIdx.x;
    if (idx >= (long)N_STOCK * NM) return;
    int n = (int)(idx / NM), m = (int)(idx % NM);
    float s = 0.f;
    if (m < 2) { for (int h = 0; h < NHEAD; ++h) s += ascore[n * 16 + h * 2 + m]; }
    else       { for (int h = 0; h < NHEAD; ++h) s += scores[(long)n * SC_LD + h * N_STOCK + (m - 2)]; }
    dst[idx] = s * 0.125f;
}

__global__ void attnmean_kernel(const float* __restrict__ attnA,
                                const __bf16* __restrict__ attnS,
                                float* __restrict__ dst)
{
    long idx = (long)blockIdx.x * blockDim.x + threadIdx.x;
    if (idx >= (long)N_STOCK * NM) return;
    int n = (int)(idx / NM), m = (int)(idx % NM);
    float s = 0.f;
    if (m < 2) { for (int h = 0; h < NHEAD; ++h) s += attnA[n * 16 + h * 2 + m]; }
    else       { for (int h = 0; h < NHEAD; ++h) s += (float)attnS[(long)n * SC_LD + h * N_STOCK + (m - 2)]; }
    dst[idx] = s * 0.125f;
}

__global__ void anchor_add_kernel(const float* __restrict__ attnA,
                                  const float* __restrict__ vA,
                                  float* __restrict__ out_pre,
                                  __bf16* __restrict__ out_preb)
{
    long idx = (long)blockIdx.x * blockDim.x + threadIdx.x;
    if (idx >= (long)N_STOCK * DIM) return;
    int n = (int)(idx >> 9), c = (int)(idx & (DIM - 1)), h = c >> 6;
    float v = out_pre[idx]
            + attnA[n * 16 + h * 2 + 0] * vA[c]
            + attnA[n * 16 + h * 2 + 1] * vA[DIM + c];
    out_pre[idx]  = v;
    out_preb[idx] = (__bf16)v;
}

__global__ void reduce_partial_kernel(const float* __restrict__ x, long n,
                                      float* __restrict__ partial)
{
    __shared__ float red[256];
    int tid = threadIdx.x;
    float s = 0.f;
    for (long i = (long)blockIdx.x * 256 + tid; i < n; i += (long)gridDim.x * 256) s += x[i];
    red[tid] = s;
    __syncthreads();
    for (int off = 128; off > 0; off >>= 1) {
        if (tid < off) red[tid] += red[tid + off];
        __syncthreads();
    }
    if (tid == 0) partial[blockIdx.x] = red[0];
}

__global__ void gatemean_final_kernel(const float* __restrict__ partial, float* __restrict__ dst)
{
    __shared__ float red[256];
    int tid = threadIdx.x;
    red[tid] = partial[tid];
    __syncthreads();
    for (int off = 128; off > 0; off >>= 1) {
        if (tid < off) red[tid] += red[tid + off];
        __syncthreads();
    }
    if (tid == 0) dst[0] = red[0] / ((float)N_STOCK * (float)DIM);
}

__global__ void __launch_bounds__(256)
final_kernel(const float* __restrict__ stock, const float* __restrict__ gate,
             const float* __restrict__ outp, const float* __restrict__ norm_w,
             float* __restrict__ dst)
{
    __shared__ float red[256];
    const int n = blockIdx.x, tid = threadIdx.x;
    const long base = (long)n * DIM;
    float g0 = stock[base + tid]       + gate[base + tid]       * outp[base + tid];
    float g1 = stock[base + tid + 256] + gate[base + tid + 256] * outp[base + tid + 256];
    red[tid] = g0 * g0 + g1 * g1;
    __syncthreads();
    for (int off = 128; off > 0; off >>= 1) {
        if (tid < off) red[tid] += red[tid + off];
        __syncthreads();
    }
    const float rms = rsqrtf(red[0] / (float)DIM + 1e-6f);
    dst[base + tid]       = g0 * rms * norm_w[tid];
    dst[base + tid + 256] = g1 * rms * norm_w[tid + 256];
}

// ---------------------------------------------------------------------------
extern "C" void kernel_launch(void* const* d_in, const int* in_sizes, int n_in,
                              void* d_out, int out_size, void* d_ws, size_t ws_size,
                              hipStream_t stream)
{
    const float* stock  = (const float*)d_in[0];
    const float* z1     = (const float*)d_in[1];
    const float* z2     = (const float*)d_in[2];
    const float* Wq     = (const float*)d_in[3];
    const float* Wk     = (const float*)d_in[4];
    const float* Wv     = (const float*)d_in[5];
    const float* Wg     = (const float*)d_in[6];
    const float* bg     = (const float*)d_in[7];
    const float* Wag    = (const float*)d_in[8];
    const float* bag    = (const float*)d_in[9];
    const float* norm_w = (const float*)d_in[10];
    const float* Wo     = (const float*)d_in[11];
    const float* bo     = (const float*)d_in[12];

    char* wp = (char*)d_ws;
    #define CARVE(T, name, elems) \
        T* name = (T*)wp; wp += (((size_t)(elems) * sizeof(T)) + 255) & ~(size_t)255;
    CARVE(float,  regime,   DIM)
    CARVE(__bf16, stockb,   (size_t)N_STOCK * DIM)
    CARVE(__bf16, Qb,       (size_t)N_STOCK * DIM)
    CARVE(__bf16, Kb,       (size_t)N_STOCK * DIM)
    CARVE(__bf16, KbT,      (size_t)N_STOCK * DIM)   // [h*64+k][m], ld = 4096
    CARVE(__bf16, Vb,       (size_t)N_STOCK * DIM)
    CARVE(float,  kA,       2 * DIM)
    CARVE(float,  vA,       2 * DIM)
    CARVE(float,  ascore,   (size_t)N_STOCK * NHEAD * 2)
    CARVE(float,  scores,   (size_t)N_STOCK * NHEAD * N_STOCK)   // 512 MB
    CARVE(float,  attnA,    (size_t)N_STOCK * NHEAD * 2)
    CARVE(__bf16, attnS,    (size_t)N_STOCK * NHEAD * N_STOCK)   // 256 MB
    CARVE(float,  out_pre,  (size_t)N_STOCK * DIM)
    CARVE(__bf16, out_preb, (size_t)N_STOCK * DIM)
    CARVE(float,  outf,     (size_t)N_STOCK * DIM)
    CARVE(__bf16, outb,     (size_t)N_STOCK * DIM)
    CARVE(float,  gatef,    (size_t)N_STOCK * DIM)
    CARVE(float,  partial,  256)
    #undef CARVE

    float* o_final = (float*)d_out;
    float* o_raw   = o_final + (size_t)N_STOCK * DIM;
    float* o_gmean = o_raw + (size_t)N_STOCK * NM;
    float* o_amean = o_gmean + 1;

    const float scale = 0.125f;   // 1/sqrt(64)

    regime_kernel<<<1, DIM, 0, stream>>>(z1, z2, Wg, bg, regime);
    {
        long n = (long)N_STOCK * DIM;
        cast_bf16_kernel<<<(int)((n + 255) / 256), 256, 0, stream>>>(stock, stockb, n);
    }
    // Q, K(+transposed copy), V projections
    dim3 gProj(DIM / 64, N_STOCK / 64, 1);
    wmma_gemm<1, true, true, false><<<gProj, 128, 0, stream>>>(stockb, DIM, nullptr, 0, DIM,
        Wq, DIM, nullptr, Qb, DIM, nullptr, 0, DIM, regime, 1.f, 0, 0, 0);
    wmma_gemm<5, true, true, false><<<gProj, 128, 0, stream>>>(stockb, DIM, nullptr, 0, DIM,
        Wk, DIM, nullptr, Kb, DIM, KbT, N_STOCK, DIM, nullptr, 1.f, 0, 0, 0);
    wmma_gemm<2, true, true, false><<<gProj, 128, 0, stream>>>(stockb, DIM, nullptr, 0, DIM,
        Wv, DIM, nullptr, Vb, DIM, nullptr, 0, DIM, nullptr, 1.f, 0, 0, 0);

    anchor_kv_kernel<<<4, 256, 0, stream>>>(z1, z2, Wk, Wv, kA, vA);
    anchor_scores_kernel<<<(N_STOCK * NHEAD * 2) / 256, 256, 0, stream>>>(Qb, kA, ascore);

    // scores[n,h,m] = scale * Qh . KbT_h   (NN via pre-transposed K, TDM-staged)
    {
        dim3 g(N_STOCK / 64, N_STOCK / 64, NHEAD);
        wmma_gemm<0, false, false, false><<<g, 128, 0, stream>>>(
            Qb, DIM, nullptr, 0, HDIM,
            KbT, N_STOCK, scores, nullptr, SC_LD, nullptr, 0, HDIM, nullptr, scale,
            /*hA=*/HDIM, /*hB=*/(long)HDIM * N_STOCK, /*hC=*/N_STOCK);
    }
    {
        long n = (long)N_STOCK * NM;
        rawscores_kernel<<<(int)((n + 255) / 256), 256, 0, stream>>>(scores, ascore, o_raw);
    }
    topk_softmax_kernel<<<N_STOCK * NHEAD, 256, 0, stream>>>(scores, ascore, attnA, attnS);
    {
        long n = (long)N_STOCK * NM;
        attnmean_kernel<<<(int)((n + 255) / 256), 256, 0, stream>>>(attnA, attnS, o_amean);
    }
    // attn @ V  (NN, K = 4096, TDM-staged)
    {
        dim3 g(1, N_STOCK / 64, NHEAD);
        wmma_gemm<0, false, false, false><<<g, 128, 0, stream>>>(
            attnS, SC_LD, nullptr, 0, N_STOCK,
            Vb, DIM, out_pre, nullptr, DIM, nullptr, 0, N_STOCK, nullptr, 1.f,
            /*hA=*/N_STOCK, /*hB=*/HDIM, /*hC=*/HDIM);
    }
    {
        long n = (long)N_STOCK * DIM;
        anchor_add_kernel<<<(int)((n + 255) / 256), 256, 0, stream>>>(attnA, vA, out_pre, out_preb);
    }
    wmma_gemm<3, true, true, false><<<gProj, 128, 0, stream>>>(out_preb, DIM, nullptr, 0, DIM,
        Wo, DIM, outf, outb, DIM, nullptr, 0, DIM, bo, 1.f, 0, 0, 0);
    wmma_gemm<4, true, true, true><<<gProj, 128, 0, stream>>>(stockb, DIM, outb, DIM, DIM,
        Wag, 2 * DIM, gatef, nullptr, DIM, nullptr, 0, 2 * DIM, bag, 1.f, 0, 0, 0);

    reduce_partial_kernel<<<256, 256, 0, stream>>>(gatef, (long)N_STOCK * DIM, partial);
    gatemean_final_kernel<<<1, 256, 0, stream>>>(partial, o_gmean);
    final_kernel<<<N_STOCK, 256, 0, stream>>>(stock, gatef, outf, norm_w, o_final);

    (void)in_sizes; (void)n_in; (void)out_size; (void)ws_size;
}
